// Attention_11699490914690
// MI455X (gfx1250) — matrix operand order repaired and checked by reading, not compiled, not run on hardware
//
#include <hip/hip_runtime.h>
#include <math.h>

// ---------------------------------------------------------------------------
// Shapes (compile-time constants from the reference)
// ---------------------------------------------------------------------------
#define BB 2
#define SS 1024
#define DD 1024
#define HH 16
#define DHH 64
#define SCALE_F 0.125f  // 1/sqrt(64)

typedef unsigned short u16;
typedef unsigned long long ull;
typedef __attribute__((ext_vector_type(16))) __bf16 v16bf;
typedef __attribute__((ext_vector_type(8)))  float  v8f;
typedef __attribute__((ext_vector_type(4)))  unsigned int u32x4;
typedef __attribute__((ext_vector_type(8)))  int  i32x8;
typedef __attribute__((ext_vector_type(4)))  int  i32x4;

static __device__ __forceinline__ u16 f2bf(float f) {
  unsigned u = __float_as_uint(f);
  u += 0x7FFFu + ((u >> 16) & 1u);  // RNE
  return (u16)(u >> 16);
}

union FragBF {
  v16bf v;
  u16   u[16];
  uint4 q[2];
};

// Generic pointer to an LDS object: low 32 bits are the LDS byte offset
// (flat aperture decode, ISA 10.2).
static __device__ __forceinline__ unsigned lds_off_u32(const void* p) {
  return (unsigned)(ull)p;
}

static __device__ __forceinline__ void async_b128(unsigned ldsOff,
                                                  const void* g) {
#if defined(__gfx1250__)
  asm volatile("global_load_async_to_lds_b128 %0, %1, off"
               :: "v"(ldsOff), "v"(g) : "memory");
#else
  (void)ldsOff; (void)g;
#endif
}
static __device__ __forceinline__ void wait_async_le4() {
#if defined(__gfx1250__)
  asm volatile("s_wait_asynccnt 0x4" ::: "memory");
#endif
}
static __device__ __forceinline__ void wait_async_le0() {
#if defined(__gfx1250__)
  asm volatile("s_wait_asynccnt 0x0" ::: "memory");
#endif
}

// ---------------------------------------------------------------------------
// Generic batched NT GEMM: C[M,N] = alpha * A[M,K] * B[N,K]^T  (bf16 in, f32 out)
// Block = 128 threads = 4 waves computing a 64x64 C tile (16x64 per wave).
// The shared 64-column x 64-K B slice (8KB) is double-buffered in LDS via
// GLOBAL_LOAD_ASYNC_TO_LDS_B128: 4 straight-line async issues per thread per
// stage (ASYNCcnt=4/wave), drained with s_wait_asynccnt. 8 WMMAs per barrier
// period. Requires M%64==0, N%64==0, K%64==0 (true at every call site).
// ---------------------------------------------------------------------------
__global__ void __launch_bounds__(128)
gemm_nt_bf16(const u16* __restrict__ A, const u16* __restrict__ B,
             float* __restrict__ C, int M, int N, int K,
             int lda, int ldb, int ldc,
             ull aBatch, ull bBatch, ull cOuter, ull cInner,
             int innerCount, float alpha) {
  __shared__ __align__(16) u16 blds[2][64][64];  // 16KB double-buffered B

  const int z = blockIdx.z;
  A += (ull)z * aBatch;
  B += (ull)z * bBatch;
  C += (ull)(z / innerCount) * cOuter + (ull)(z % innerCount) * cInner;

  const int tid  = threadIdx.x;
  const int lane = tid & 31;
  const int wave = tid >> 5;
  const int rowBase = blockIdx.y * 64 + wave * 16;
  const int colBase = blockIdx.x * 64;

  const int h2 = lane >> 4;  // lane half select
  const int r  = lane & 15;

  // Per-thread async chunk bases (chunk c: B row colBase+(c>>3), halves (c&7)*8,
  // LDS linear byte c*16). This thread owns chunks tid, tid+128, +256, +384.
  const unsigned ldsBase = lds_off_u32(&blds[0][0][0]);
  const u16* gB0 = B + (ull)(colBase + ((tid      ) >> 3)) * ldb + ((tid      ) & 7) * 8;
  const u16* gB1 = B + (ull)(colBase + ((tid + 128) >> 3)) * ldb + ((tid + 128) & 7) * 8;
  const u16* gB2 = B + (ull)(colBase + ((tid + 256) >> 3)) * ldb + ((tid + 256) & 7) * 8;
  const u16* gB3 = B + (ull)(colBase + ((tid + 384) >> 3)) * ldb + ((tid + 384) & 7) * 8;
  const unsigned l0 = ldsBase + (unsigned)tid * 16;

  auto stage = [&](int buf, int kk) {
    const unsigned lb = l0 + (unsigned)buf * 8192u;
#if defined(__gfx1250__)
    async_b128(lb +    0u, gB0 + kk);
    async_b128(lb + 2048u, gB1 + kk);
    async_b128(lb + 4096u, gB2 + kk);
    async_b128(lb + 6144u, gB3 + kk);
#else
    char* base = (char*)&blds[0][0][0] + buf * 8192 + tid * 16;
    *(uint4*)(base +    0) = *(const uint4*)(gB0 + kk);
    *(uint4*)(base + 2048) = *(const uint4*)(gB1 + kk);
    *(uint4*)(base + 4096) = *(const uint4*)(gB2 + kk);
    *(uint4*)(base + 6144) = *(const uint4*)(gB3 + kk);
#endif
  };

  // A fragment base: lane holds row (rowBase+r); K run of 16 at h2*16.
  const u16* aRow = A + (ull)(rowBase + r) * lda + h2 * 16;

  v8f acc0 = {}, acc1 = {}, acc2 = {}, acc3 = {};

  auto period = [&](int buf, int k) {
    const u16(*bt)[64] = blds[buf];
#pragma unroll
    for (int s2 = 0; s2 < 64; s2 += 32) {
      FragBF a;
      {
        const uint4* pa = (const uint4*)(aRow + k + s2);
        a.q[0] = pa[0];
        a.q[1] = pa[1];  // +16B = rest of the contiguous K run
      }
      FragBF b0, b1, b2, b3;
      {  // B fragment: lane = column (t*16+r); 32B at halves s2 + h2*16
        const uint4* q0 = (const uint4*)&bt[ 0 + r][s2 + h2 * 16];
        const uint4* q1 = (const uint4*)&bt[16 + r][s2 + h2 * 16];
        const uint4* q2 = (const uint4*)&bt[32 + r][s2 + h2 * 16];
        const uint4* q3 = (const uint4*)&bt[48 + r][s2 + h2 * 16];
        b0.q[0] = q0[0]; b0.q[1] = q0[1];
        b1.q[0] = q1[0]; b1.q[1] = q1[1];
        b2.q[0] = q2[0]; b2.q[1] = q2[1];
        b3.q[0] = q3[0]; b3.q[1] = q3[1];
      }
      acc0 = __builtin_amdgcn_wmma_f32_16x16x32_bf16(false, a.v, false, b0.v,
                                                     (short)0, acc0, false, false);
      acc1 = __builtin_amdgcn_wmma_f32_16x16x32_bf16(false, a.v, false, b1.v,
                                                     (short)0, acc1, false, false);
      acc2 = __builtin_amdgcn_wmma_f32_16x16x32_bf16(false, a.v, false, b2.v,
                                                     (short)0, acc2, false, false);
      acc3 = __builtin_amdgcn_wmma_f32_16x16x32_bf16(false, a.v, false, b3.v,
                                                     (short)0, acc3, false, false);
    }
  };

  stage(0, 0);
  int buf = 0, k = 0;
  for (; k + 64 < K; k += 64) {        // steady state: always stage next
    stage(buf ^ 1, k + 64);
    wait_async_le4();                   // previous stage has landed
    __syncthreads();
    __builtin_prefetch(aRow + k + 256, 0, 3);
    period(buf, k);
    __syncthreads();                    // reads done before overwrite
    buf ^= 1;
  }
  wait_async_le0();                     // tail
  __syncthreads();
  period(buf, k);

  // C/D layout: row = v + 8*h2, col = lane&15
  const int crow = rowBase + h2 * 8;
  const int ccol = colBase + r;
  for (int v = 0; v < 8; ++v) {
    float* cp = C + (ull)(crow + v) * ldc;
    cp[ccol +  0] = alpha * acc0[v];
    cp[ccol + 16] = alpha * acc1[v];
    cp[ccol + 32] = alpha * acc2[v];
    cp[ccol + 48] = alpha * acc3[v];
  }
}

// ---------------------------------------------------------------------------
// f32 -> bf16 cast
// ---------------------------------------------------------------------------
__global__ void __launch_bounds__(256)
cast_f32_bf16(const float* __restrict__ in, u16* __restrict__ out, int n) {
  int i = blockIdx.x * 256 + threadIdx.x;
  if (i < n) out[i] = f2bf(in[i]);
}

// ---------------------------------------------------------------------------
// qkv [B*S, 3*H*DH] f32 -> q,k [B,H,S,DH] bf16 and vT [B,H,DH,S] bf16
// ---------------------------------------------------------------------------
__global__ void __launch_bounds__(256)
reorg_qkv(const float* __restrict__ qkv, u16* __restrict__ qb,
          u16* __restrict__ kb, u16* __restrict__ vT) {
  int i = blockIdx.x * 256 + threadIdx.x;
  if (i >= BB * SS * HH * DHH) return;
  int d = i & 63;
  int h = (i >> 6) & 15;
  int s = (i >> 10) & 1023;
  int b = i >> 20;
  ull row = (ull)(b * SS + s) * (3 * HH * DHH);
  int col = h * DHH + d;
  float q = qkv[row + col];
  float k = qkv[row + HH * DHH + col];
  float v = qkv[row + 2 * HH * DHH + col];
  ull bh = (ull)b * HH + h;
  qb[(bh * SS + s) * DHH + d] = f2bf(q);
  kb[(bh * SS + s) * DHH + d] = f2bf(k);
  vT[(bh * DHH + d) * SS + s] = f2bf(v);
}

// ---------------------------------------------------------------------------
// Fused Kerple bias + DAPE MLP (both layers as WMMA) + residual add, in-place
// on scores. One block = one batch b and a 16x16 (i,j) tile = 256 pairs.
// Phase 1 input (16 heads x 16x16 f32 = 16KB, a 3-D strided tile) is fetched
// with ONE Tensor Data Mover DMA per block:
//   x=j (16 contig), y=i (stride S), z=h (stride S*S), data_size=4B.
// Layer 1: Z[256,32] @ W1[16,32]^T  (K=32, one WMMA per 16-row tile)
// Layer 2: H[256,16] @ W2[16,16]^T  (K padded to 32 with zeros)
// ---------------------------------------------------------------------------
__global__ void __launch_bounds__(256)
dape_kernel(float* __restrict__ scores, const float* __restrict__ bias_p,
            const float* __restrict__ bias_a, const float* __restrict__ w1,
            const float* __restrict__ b1v, const float* __restrict__ w2,
            const float* __restrict__ b2v) {
  __shared__ __align__(16) float sraw[HH][16][16];  // TDM landing: [h][i][j]
  __shared__ __align__(16) u16   zlds[256][32];     // [pair][feature] bf16
  __shared__ __align__(16) u16   hlds[256][32];     // hidden, K-padded zeros
  __shared__ float slds[256][16];                   // scores + kb in f32

  const int b = blockIdx.z;
  const int iBase = blockIdx.y * 16;
  const int jBase = blockIdx.x * 16;
  const int p = threadIdx.x;  // pair index in tile
  const int ii = p >> 4, jj = p & 15;
  const float dist = fabsf((float)((iBase + ii) - (jBase + jj)));

  // ---- Phase 0: TDM DMA of the score tile into LDS --------------------
  ull tileElem = ((ull)b * HH * SS + iBase) * SS + jBase;  // h=0 element
#if defined(__gfx1250__)
  if (threadIdx.x < 32) {  // one wave issues the DMA (EXEC ignored by TDM)
    ull gaddr = (ull)(scores + tileElem);
    u32x4 g0;
    g0.x = 1u;                              // count=1, user mode
    g0.y = lds_off_u32(&sraw[0][0][0]);     // lds_addr
    g0.z = (unsigned)gaddr;                 // global_addr[31:0]
    g0.w = ((unsigned)(gaddr >> 32) & 0x01FFFFFFu) | (2u << 30);  // type=2
    i32x8 g1;
    g1[0] = 0x00020000;             // data_size=4B, no mask/flags
    g1[1] = (int)(SS << 16);        // tensor_dim0[15:0] in [31:16]
    g1[2] = (int)(SS << 16);        // dim0[31:16]=0 | tensor_dim1[15:0]
    g1[3] = (int)(16 << 16);        // dim1[31:16]=0 | tile_dim0=16
    g1[4] = (int)((16 << 16) | 16); // tile_dim1=16 | tile_dim2=16
    g1[5] = (int)SS;                // tensor_dim0_stride[31:0] = 1024
    g1[6] = 0;                      // stride0[47:32]=0 | stride1[15:0]=0
    g1[7] = (int)(((ull)SS * SS) >> 16);  // stride1[47:16] = 16
    i32x4 g2;
    g2[0] = HH; g2[1] = 0; g2[2] = 0; g2[3] = 0;  // tensor_dim2 = 16 heads
    i32x4 g3;
    g3[0] = 0; g3[1] = 0; g3[2] = 0; g3[3] = 0;
#if __has_builtin(__builtin_amdgcn_tensor_load_to_lds)
#if __clang_major__ >= 23
    i32x8 gpad;
    gpad[0]=0; gpad[1]=0; gpad[2]=0; gpad[3]=0;
    gpad[4]=0; gpad[5]=0; gpad[6]=0; gpad[7]=0;
    __builtin_amdgcn_tensor_load_to_lds(g0, g1, g2, g3, gpad, 0);
#else
    __builtin_amdgcn_tensor_load_to_lds(g0, g1, g2, g3, 0);
#endif
    __builtin_amdgcn_s_wait_tensorcnt((short)0);
#else
    asm volatile("tensor_load_to_lds %0, %1, %2, %3"
                 :: "s"(g0), "s"(g1), "s"(g2), "s"(g3) : "memory");
    asm volatile("s_wait_tensorcnt 0x0" ::: "memory");
#endif
  }
  __syncthreads();  // DMA drained by wave 0; LDS tile visible to all
#endif

  // ---- Phase 1: build Z features + f32 score+kb copy ------------------
  for (int h = 0; h < HH; ++h) {
#if defined(__gfx1250__)
    float s = sraw[h][ii][jj];
#else
    float s = scores[tileElem + (ull)h * SS * SS + (ull)ii * SS + jj];
#endif
    float pc = fmaxf(bias_p[h], 0.01f);
    float ac = fmaxf(bias_a[h], 0.01f);
    float kb = -pc * log1pf(ac * dist);
    zlds[p][h]      = f2bf(s);
    zlds[p][HH + h] = f2bf(kb);
    slds[p][h]      = s + kb;
    hlds[p][HH + h] = 0;  // zero pad for layer-2 K
  }
  __syncthreads();

  const int lane = threadIdx.x & 31;
  const int wave = threadIdx.x >> 5;
  const int h2 = lane >> 4, r = lane & 15;

  // Weight fragments (B-matrix layout [N,K]): lane holds output column r.
  FragBF fw1, fw2;
  {
    const int kb2 = h2 * 16;
    for (int e = 0; e < 16; ++e) fw1.u[e] = f2bf(w1[r * 32 + kb2 + e]);
    for (int e = 0; e < 16; ++e)
      fw2.u[e] = (h2 == 0) ? f2bf(w2[r * 16 + e]) : (u16)0;  // K>=16 padding
  }
  const float bb1 = b1v[r];
  const float bb2 = b2v[r];

  // 8 waves x 2 sub-tiles cover all 16 M-tiles (256 pairs)
#pragma unroll
  for (int sub = 0; sub < 2; ++sub) {
    const int mBase = (wave * 2 + sub) * 16;

    FragBF a;
    {
      const uint4* pz = (const uint4*)&zlds[mBase + r][h2 * 16];
      a.q[0] = pz[0];
      a.q[1] = pz[1];
    }
    v8f c1 = {};
    c1 = __builtin_amdgcn_wmma_f32_16x16x32_bf16(false, a.v, false, fw1.v,
                                                 (short)0, c1, false, false);
    for (int v = 0; v < 8; ++v) {
      float x = c1[v] + bb1;
      float g = 0.5f * x * (1.0f + erff(x * 0.70710678118f));  // exact GELU
      hlds[mBase + v + 8 * h2][r] = f2bf(g);
    }

    FragBF a2;
    {
      const uint4* ph = (const uint4*)&hlds[mBase + r][h2 * 16];
      a2.q[0] = ph[0];
      a2.q[1] = ph[1];
    }
    v8f c2 = {};
    c2 = __builtin_amdgcn_wmma_f32_16x16x32_bf16(false, a2.v, false, fw2.v,
                                                 (short)0, c2, false, false);
    for (int v = 0; v < 8; ++v) {
      int pr = mBase + v + 8 * h2;
      float outv = slds[pr][r] + c2[v] + bb2;
      int oi = iBase + (pr >> 4);
      int oj = jBase + (pr & 15);
      scores[((((ull)b * HH + r) * SS) + oi) * SS + oj] = outv;
    }
  }
}

// ---------------------------------------------------------------------------
// Row softmax over last dim (1024) of scores; output bf16 attn.
// ---------------------------------------------------------------------------
__global__ void __launch_bounds__(256)
softmax_kernel(const float* __restrict__ scores, u16* __restrict__ attn) {
  __shared__ float red[256];
  const ull row = blockIdx.x;
  const float* src = scores + row * SS;
  u16* dst = attn + row * SS;
  const int t = threadIdx.x;

  float x0 = src[t], x1 = src[t + 256], x2 = src[t + 512], x3 = src[t + 768];
  float m = fmaxf(fmaxf(x0, x1), fmaxf(x2, x3));
  red[t] = m;
  __syncthreads();
  for (int s = 128; s > 0; s >>= 1) {
    if (t < s) red[t] = fmaxf(red[t], red[t + s]);
    __syncthreads();
  }
  m = red[0];
  __syncthreads();
  float e0 = __expf(x0 - m), e1 = __expf(x1 - m);
  float e2 = __expf(x2 - m), e3 = __expf(x3 - m);
  red[t] = e0 + e1 + e2 + e3;
  __syncthreads();
  for (int s = 128; s > 0; s >>= 1) {
    if (t < s) red[t] += red[t + s];
    __syncthreads();
  }
  float inv = 1.0f / red[0];
  dst[t]       = f2bf(e0 * inv);
  dst[t + 256] = f2bf(e1 * inv);
  dst[t + 512] = f2bf(e2 * inv);
  dst[t + 768] = f2bf(e3 * inv);
}

// ---------------------------------------------------------------------------
// Host-side orchestration
// ---------------------------------------------------------------------------
extern "C" void kernel_launch(void* const* d_in, const int* in_sizes, int n_in,
                              void* d_out, int out_size, void* d_ws,
                              size_t ws_size, hipStream_t stream) {
  const float* x      = (const float*)d_in[0];
  const float* qkv_w  = (const float*)d_in[1];
  const float* out_w  = (const float*)d_in[2];
  const float* bias_p = (const float*)d_in[3];
  const float* bias_a = (const float*)d_in[4];
  const float* mlp_w1 = (const float*)d_in[5];
  const float* mlp_b1 = (const float*)d_in[6];
  const float* mlp_w2 = (const float*)d_in[7];
  const float* mlp_b2 = (const float*)d_in[8];
  float* out = (float*)d_out;

  char* ws = (char*)d_ws;
  size_t off = 0;
  auto alloc = [&](size_t bytes) -> char* {
    char* p = ws + off;
    off += (bytes + 255) & ~(size_t)255;
    return p;
  };

  const int M = BB * SS;  // 2048 tokens
  u16*   x_bf    = (u16*)alloc((size_t)M * DD * 2);
  u16*   wqkv_bf = (u16*)alloc((size_t)3 * HH * DHH * DD * 2);
  u16*   wout_bf = (u16*)alloc((size_t)DD * HH * DHH * 2);
  float* qkv_f   = (float*)alloc((size_t)M * 3 * HH * DHH * 4);
  u16*   q_bf    = (u16*)alloc((size_t)BB * HH * SS * DHH * 2);
  u16*   k_bf    = (u16*)alloc((size_t)BB * HH * SS * DHH * 2);
  u16*   vT_bf   = (u16*)alloc((size_t)BB * HH * DHH * SS * 2);
  float* scores  = (float*)alloc((size_t)BB * HH * SS * SS * 4);
  u16*   attn_bf = (u16*)alloc((size_t)BB * HH * SS * SS * 2);
  float* aout_f  = (float*)alloc((size_t)M * HH * DHH * 4);
  u16*   aout_bf = (u16*)alloc((size_t)M * HH * DHH * 2);
  (void)ws_size; (void)n_in; (void)in_sizes; (void)out_size;

  // 1) casts to bf16
  {
    int n = M * DD;
    cast_f32_bf16<<<(n + 255) / 256, 256, 0, stream>>>(x, x_bf, n);
    n = 3 * HH * DHH * DD;
    cast_f32_bf16<<<(n + 255) / 256, 256, 0, stream>>>(qkv_w, wqkv_bf, n);
    n = DD * HH * DHH;
    cast_f32_bf16<<<(n + 255) / 256, 256, 0, stream>>>(out_w, wout_bf, n);
  }

  // 2) QKV projection: [2048,1024] x [3072,1024]^T -> [2048,3072]
  gemm_nt_bf16<<<dim3(3 * HH * DHH / 64, M / 64, 1), 128, 0, stream>>>(
      x_bf, wqkv_bf, qkv_f, M, 3 * HH * DHH, DD, DD, DD, 3 * HH * DHH,
      0ull, 0ull, 0ull, 0ull, 1, 1.0f);

  // 3) split/transpose into q,k [B,H,S,DH] and vT [B,H,DH,S]
  {
    int n = BB * SS * HH * DHH;
    reorg_qkv<<<(n + 255) / 256, 256, 0, stream>>>(qkv_f, q_bf, k_bf, vT_bf);
  }

  // 4) scores = SCALE * q k^T per (b,h): batched M=N=1024, K=64
  gemm_nt_bf16<<<dim3(SS / 64, SS / 64, BB * HH), 128, 0, stream>>>(
      q_bf, k_bf, scores, SS, SS, DHH, DHH, DHH, SS,
      (ull)SS * DHH, (ull)SS * DHH, (ull)SS * SS, 0ull, 1, SCALE_F);

  // 5) fused kerple bias + DAPE MLP + residual, in place on scores
  dape_kernel<<<dim3(SS / 16, SS / 16, BB), 256, 0, stream>>>(
      scores, bias_p, bias_a, mlp_w1, mlp_b1, mlp_w2, mlp_b2);

  // 6) softmax rows -> bf16 attn
  softmax_kernel<<<dim3(BB * HH * SS), 256, 0, stream>>>(scores, attn_bf);

  // 7) attn @ v per (b,h): A=[1024,1024] attn, B=vT [64,1024] -> C strip into
  //    [B,S,H*DH] at column offset h*64
  gemm_nt_bf16<<<dim3(DHH / 64, SS / 64, BB * HH), 128, 0, stream>>>(
      attn_bf, vT_bf, aout_f, SS, DHH, SS, SS, SS, HH * DHH,
      (ull)SS * SS, (ull)DHH * SS,
      (ull)SS * HH * DHH, (ull)DHH, HH, 1.0f);

  // 8) cast attention output to bf16
  {
    int n = M * HH * DHH;
    cast_f32_bf16<<<(n + 255) / 256, 256, 0, stream>>>(aout_f, aout_bf, n);
  }

  // 9) output projection: [2048,1024] x [1024,1024]^T -> d_out
  gemm_nt_bf16<<<dim3(DD / 64, M / 64, 1), 128, 0, stream>>>(
      aout_bf, wout_bf, out, M, DD, HH * DHH, HH * DHH, HH * DHH, DD,
      0ull, 0ull, 0ull, 0ull, 1, 1.0f);
}
